// DSSM_18391049962085
// MI455X (gfx1250) — matrix operand order repaired and checked by reading, not compiled, not run on hardware
//
#include <hip/hip_runtime.h>
#include <math.h>

// Problem constants (match reference)
#define B_  4
#define C_  96
#define H_  64
#define W_  64
#define L_  4096      // H_*W_
#define K_  3
#define NS_ 16
#define R_  6
#define NR_ 38        // R_ + 2*NS_

typedef __attribute__((ext_vector_type(16))) _Float16 v16h;
typedef __attribute__((ext_vector_type(8)))  float    v8f;

__device__ __forceinline__ float gelu_exact(float x) {
  return 0.5f * x * (1.f + erff(x * 0.70710678118654752f));
}
__device__ __forceinline__ float softplus_f(float x) {
  return (x > 20.f) ? x : log1pf(__expf(x));
}
__device__ __forceinline__ float sigmoid_f(float x) {
  return 1.f / (1.f + __expf(-x));
}

// ---------------- conv_offset pipeline ----------------

// Channel mean: one block per (b,c)
__global__ void k_mean(const float* __restrict__ x, float* __restrict__ xc) {
  __shared__ float red[256];
  size_t base = (size_t)blockIdx.x * L_;
  float s = 0.f;
  for (int i = threadIdx.x; i < L_; i += 256) s += x[base + i];
  red[threadIdx.x] = s;
  __syncthreads();
  for (int st = 128; st > 0; st >>= 1) {
    if ((int)threadIdx.x < st) red[threadIdx.x] += red[threadIdx.x + st];
    __syncthreads();
  }
  if (threadIdx.x == 0) xc[blockIdx.x] = red[0] * (1.f / (float)L_);
}

// Channel attention: ca[b,c] = sigmoid(gelu(xc @ W1^T + b1) @ W2^T + b2)
__global__ void k_ca(const float* __restrict__ xc,
                     const float* __restrict__ W1, const float* __restrict__ bb1,
                     const float* __restrict__ W2, const float* __restrict__ bb2,
                     float* __restrict__ ca) {
  int idx = blockIdx.x * blockDim.x + threadIdx.x;
  if (idx >= B_ * C_) return;
  int b = idx / C_, c = idx % C_;
  float acc = bb2[c];
  for (int h = 0; h < C_ / 16; ++h) {
    float hv = bb1[h];
    for (int d = 0; d < C_; ++d) hv += xc[b * C_ + d] * W1[h * C_ + d];
    acc += gelu_exact(hv) * W2[c * (C_ / 16) + h];
  }
  ca[idx] = sigmoid_f(acc);
}

// Depthwise 9x9 conv (pad 4) + bias, then * ca
__global__ void k_dwconv(const float* __restrict__ x, const float* __restrict__ w,
                         const float* __restrict__ bias, const float* __restrict__ ca,
                         float* __restrict__ x1) {
  int idx = blockIdx.x * blockDim.x + threadIdx.x;
  if (idx >= B_ * C_ * L_) return;
  int p = idx % L_;
  int bc = idx / L_;
  int c = bc % C_;
  int i = p / W_, j = p % W_;
  float acc = bias[c];
  const float* xb = x + (size_t)bc * L_;
  const float* wc = w + c * 81;
  for (int u = 0; u < 9; ++u) {
    int yy = i + u - 4;
    if (yy < 0 || yy >= H_) continue;
    for (int v = 0; v < 9; ++v) {
      int xx = j + v - 4;
      if (xx < 0 || xx >= W_) continue;
      acc += xb[yy * W_ + xx] * wc[u * 9 + v];
    }
  }
  x1[idx] = acc * ca[bc];
}

// Per-pixel LayerNorm over C, GELU, then 1x1 conv C->3 -> off (B,3,L)
__global__ void k_lnoff(const float* __restrict__ x1, const float* __restrict__ g,
                        const float* __restrict__ bt, const float* __restrict__ w2,
                        float* __restrict__ off) {
  int idx = blockIdx.x * blockDim.x + threadIdx.x;
  if (idx >= B_ * L_) return;
  int b = idx / L_, p = idx % L_;
  const float* xb = x1 + (size_t)b * C_ * L_ + p;
  float mean = 0.f;
  for (int c = 0; c < C_; ++c) mean += xb[(size_t)c * L_];
  mean *= (1.f / C_);
  float var = 0.f;
  for (int c = 0; c < C_; ++c) { float d = xb[(size_t)c * L_] - mean; var += d * d; }
  var *= (1.f / C_);
  float inv = rsqrtf(var + 1e-5f);
  float a0 = 0.f, a1 = 0.f, a2 = 0.f;
  for (int c = 0; c < C_; ++c) {
    float xn = (xb[(size_t)c * L_] - mean) * inv * g[c] + bt[c];
    float xg = gelu_exact(xn);
    a0 += xg * w2[0 * C_ + c];
    a1 += xg * w2[1 * C_ + c];
    a2 += xg * w2[2 * C_ + c];
  }
  off[((size_t)b * 3 + 0) * L_ + p] = a0;
  off[((size_t)b * 3 + 1) * L_ + p] = a1;
  off[((size_t)b * 3 + 2) * L_ + p] = a2;
}

// ---------------- deformable sampling ----------------

// Resize rpe_table (C,7,7) -> (C,64,64), half-pixel centers
__global__ void k_rpe(const float* __restrict__ t, float* __restrict__ rb) {
  int idx = blockIdx.x * blockDim.x + threadIdx.x;
  if (idx >= C_ * L_) return;
  int c = idx / L_, p = idx % L_;
  int i = p / W_, j = p % W_;
  float sy = fmaxf((i + 0.5f) * (7.f / 64.f) - 0.5f, 0.f);
  float sx = fmaxf((j + 0.5f) * (7.f / 64.f) - 0.5f, 0.f);
  int y0 = (int)floorf(sy); int y1 = min(y0 + 1, 6); float wy = sy - (float)y0;
  int x0 = (int)floorf(sx); int x1 = min(x0 + 1, 6); float wx = sx - (float)x0;
  const float* tc = t + c * 49;
  float r0 = tc[y0 * 7 + x0] * (1.f - wy) + tc[y1 * 7 + x0] * wy;
  float r1 = tc[y0 * 7 + x1] * (1.f - wy) + tc[y1 * 7 + x1] * wy;
  rb[idx] = r0 * (1.f - wx) + r1 * wx;
}

__device__ __forceinline__ void mk_taps(float gx, float gy, int* ix, int* iy, float* w) {
  float fx = (gx + 1.f) * 0.5f * (W_ - 1);
  float fy = (gy + 1.f) * 0.5f * (H_ - 1);
  float x0f = floorf(fx), y0f = floorf(fy);
  float wx = fx - x0f, wy = fy - y0f;
  float ws[4] = { (1.f - wx) * (1.f - wy), wx * (1.f - wy), (1.f - wx) * wy, wx * wy };
  float xs4[4] = { x0f, x0f + 1.f, x0f, x0f + 1.f };
  float ys4[4] = { y0f, y0f, y0f + 1.f, y0f + 1.f };
#pragma unroll
  for (int t = 0; t < 4; ++t) {
    float valid = (xs4[t] >= 0.f && xs4[t] < (float)W_ && ys4[t] >= 0.f && ys4[t] < (float)H_) ? 1.f : 0.f;
    ix[t] = (int)fminf(fmaxf(xs4[t], 0.f), (float)(W_ - 1));
    iy[t] = (int)fminf(fmaxf(ys4[t], 0.f), (float)(H_ - 1));
    w[t] = ws[t] * valid;
  }
}

// xd[b,c,l] = grid_sample(x, pos) + grid_sample(rpe_bias, disp)
__global__ void k_sample(const float* __restrict__ x, const float* __restrict__ off,
                         const float* __restrict__ rb, float* __restrict__ xd) {
  int idx = blockIdx.x * blockDim.x + threadIdx.x;
  if (idx >= B_ * L_) return;
  int b = idx / L_, l = idx % L_;
  int i = l / W_, j = l % W_;
  float offy = off[((size_t)b * 3 + 0) * L_ + l];
  float offx = off[((size_t)b * 3 + 1) * L_ + l];
  float posy = tanhf(offy) * (1.f / (H_ - 1)) + ((0.5f + (float)i) / (float)(H_ - 1)) * 2.f - 1.f;
  float posx = tanhf(offx) * (1.f / (W_ - 1)) + ((0.5f + (float)j) / (float)(W_ - 1)) * 2.f - 1.f;
  float ky = ((float)i * ((float)H_ / (H_ - 1))) / (float)(H_ - 1) * 2.f - 1.f;
  float kx = ((float)j * ((float)W_ / (W_ - 1))) / (float)(W_ - 1) * 2.f - 1.f;
  float dispy = (ky - posy) * 0.5f;
  float dispx = (kx - posx) * 0.5f;
  int xix[4], xiy[4], rix[4], riy[4];
  float xw[4], rw[4];
  mk_taps(posx, posy, xix, xiy, xw);
  mk_taps(dispx, dispy, rix, riy, rw);
  for (int c = 0; c < C_; ++c) {
    const float* xb = x + ((size_t)b * C_ + c) * L_;
    const float* rc = rb + (size_t)c * L_;
    float v = 0.f;
#pragma unroll
    for (int t = 0; t < 4; ++t) v += xb[xiy[t] * W_ + xix[t]] * xw[t];
#pragma unroll
    for (int t = 0; t < 4; ++t) v += rc[riy[t] * W_ + rix[t]] * rw[t];
    xd[((size_t)b * C_ + c) * L_ + l] = v;
  }
}

// Bitonic argsort of path_pos per batch; also builds inverse permutation
__global__ __launch_bounds__(1024) void k_sort(const float* __restrict__ off,
                                               int* __restrict__ indices,
                                               int* __restrict__ inv) {
  __shared__ float key[L_];
  __shared__ int   idx[L_];
  int b = blockIdx.x;
  for (int i = threadIdx.x; i < L_; i += 1024) {
    float de = off[((size_t)b * 3 + 2) * L_ + i];
    key[i] = tanhf(de) + ((0.5f + (float)i) / (float)(L_ - 1)) * 2.f - 1.f;
    idx[i] = i;
  }
  __syncthreads();
  for (int kk = 2; kk <= L_; kk <<= 1) {
    for (int j = kk >> 1; j > 0; j >>= 1) {
      for (int i = threadIdx.x; i < L_; i += 1024) {
        int ixj = i ^ j;
        if (ixj > i) {
          bool up = ((i & kk) == 0);
          if ((key[i] > key[ixj]) == up) {
            float tk = key[i]; key[i] = key[ixj]; key[ixj] = tk;
            int   ti = idx[i]; idx[i] = idx[ixj]; idx[ixj] = ti;
          }
        }
      }
      __syncthreads();
    }
  }
  for (int i = threadIdx.x; i < L_; i += 1024) {
    indices[b * L_ + i] = idx[i];
    inv[b * L_ + idx[i]] = i;
  }
}

// xs[b,k,c,l] (f32, for scan) and xs_h[b,k,l,c] (f16 transposed, for WMMA B frags)
__global__ void k_buildxs(const float* __restrict__ x, const float* __restrict__ xd,
                          const int* __restrict__ indices, float* __restrict__ xs,
                          _Float16* __restrict__ xs_h) {
  int idx = blockIdx.x * blockDim.x + threadIdx.x;
  if (idx >= B_ * C_ * L_) return;
  int l = idx % L_;
  int bc = idx / L_;
  int c = bc % C_, b = bc / C_;
  float v0 = x[(size_t)bc * L_ + l];
  float v1 = x[(size_t)bc * L_ + (L_ - 1 - l)];
  float v2 = xd[(size_t)bc * L_ + indices[b * L_ + l]];
  size_t base = ((size_t)b * K_) * C_ * L_ + (size_t)c * L_;
  xs[base + 0 * (size_t)C_ * L_ + l] = v0;
  xs[base + 1 * (size_t)C_ * L_ + l] = v1;
  xs[base + 2 * (size_t)C_ * L_ + l] = v2;
  size_t hb = ((size_t)b * K_) * L_ * C_ + (size_t)l * C_ + c;
  xs_h[hb + 0 * (size_t)L_ * C_] = (_Float16)v0;
  xs_h[hb + 1 * (size_t)L_ * C_] = (_Float16)v1;
  xs_h[hb + 2 * (size_t)L_ * C_] = (_Float16)v2;
}

// Pre-pack x_proj_weight into f16 WMMA A-fragment order:
// Apack[k][mt][ks][lane][16], rows >= 38 zero-padded.
// A layout (16-bit A 16x32, contiguous k order used here, same as B frags): lane m = lane&15,
// element pair 2j holds K = kc + 2j (+16 for upper half-wave).
__global__ void k_packA(const float* __restrict__ Wp, _Float16* __restrict__ Apack) {
  int idx = blockIdx.x * blockDim.x + threadIdx.x;
  if (idx >= K_ * 3 * 3 * 32) return;
  int lane = idx & 31;
  int t = idx >> 5;
  int ks = t % 3; t /= 3;
  int mt = t % 3; t /= 3;
  int k = t;
  int m16 = lane & 15, half = lane >> 4;
  int row = mt * 16 + m16;
  int kc = ks * 32;
  _Float16* dst = Apack + (size_t)idx * 16;
#pragma unroll
  for (int j = 0; j < 8; ++j) {
    int ka = kc + 2 * j + half * 16;
    float a0 = (row < NR_) ? Wp[(size_t)k * NR_ * C_ + (size_t)row * C_ + ka]     : 0.f;
    float a1 = (row < NR_) ? Wp[(size_t)k * NR_ * C_ + (size_t)row * C_ + ka + 1] : 0.f;
    dst[2 * j]     = (_Float16)a0;
    dst[2 * j + 1] = (_Float16)a1;
  }
}

// ---------------- WMMA GEMM: x_dbl[b,k] = Wp[k](38x96) @ xs[b,k](96x4096) ----------------
// One wave per 16x64 output strip: 4 N-tiles share one A fragment per k-step.
// K=96 = 3 fully-unrolled steps -> 12 v_wmma_f32_16x16x32_f16 per wave.
// B fragment = 16 contiguous halves of xs_h row (32B-aligned v16h load).
__global__ __launch_bounds__(32) void k_gemm(const _Float16* __restrict__ xs_h,
                                             const _Float16* __restrict__ Apack,
                                             float* __restrict__ xdbl) {
  int blk = blockIdx.x;
  int nt4 = blk % (L_ / 64); blk /= (L_ / 64);
  int mt  = blk % 3;         blk /= 3;
  int k   = blk % K_;
  int b   = blk / K_;
  int lane = threadIdx.x & 31;
  int m16  = lane & 15;
  int half = lane >> 4;
  const _Float16* bbase = xs_h + ((size_t)(b * K_ + k)) * L_ * C_;
  v8f acc0 = {}, acc1 = {}, acc2 = {}, acc3 = {};
#pragma unroll
  for (int ks = 0; ks < 3; ++ks) {
    v16h a = *(const v16h*)(Apack + ((((size_t)k * 3 + mt) * 3 + ks) * 32 + lane) * 16);
    int coff = ks * 32 + half * 16;
    const _Float16* bp = bbase + (size_t)(nt4 * 64 + m16) * C_ + coff;
    v16h b0 = *(const v16h*)(bp + 0 * 16 * C_);
    v16h b1 = *(const v16h*)(bp + 1 * 16 * C_);
    v16h b2 = *(const v16h*)(bp + 2 * 16 * C_);
    v16h b3 = *(const v16h*)(bp + 3 * 16 * C_);
    acc0 = __builtin_amdgcn_wmma_f32_16x16x32_f16(false, a, false, b0, (short)0, acc0, false, false);
    acc1 = __builtin_amdgcn_wmma_f32_16x16x32_f16(false, a, false, b1, (short)0, acc1, false, false);
    acc2 = __builtin_amdgcn_wmma_f32_16x16x32_f16(false, a, false, b2, (short)0, acc2, false, false);
    acc3 = __builtin_amdgcn_wmma_f32_16x16x32_f16(false, a, false, b3, (short)0, acc3, false, false);
  }
  float* outb = xdbl + ((size_t)(b * K_ + k)) * NR_ * L_;
  int ncol = nt4 * 64 + m16;
#pragma unroll
  for (int r = 0; r < 8; ++r) {
    int orow = mt * 16 + r + 8 * half;   // D layout: VGPR r -> M = r + 8*half
    if (orow < NR_) {
      float* orow_p = outb + (size_t)orow * L_ + ncol;
      orow_p[0]  = acc0[r];
      orow_p[16] = acc1[r];
      orow_p[32] = acc2[r];
      orow_p[48] = acc3[r];
    }
  }
}

// ---------------- selective scan ----------------
// One workgroup (96 threads = 3 waves) per (b,k); thread d keeps h[16] in regs.
// x_dbl rows 0..5 = dt-rank, 6..21 = B, 22..37 = C. Staged in LDS per 64-col chunk.
__global__ __launch_bounds__(96) void k_scan(const float* __restrict__ xs,
                                             const float* __restrict__ xdbl,
                                             const float* __restrict__ dtw,
                                             const float* __restrict__ dtb,
                                             const float* __restrict__ Alog,
                                             const float* __restrict__ Dsp,
                                             float* __restrict__ ys) {
  __shared__ float s_xd[NR_ * 64];
  __shared__ float s_u[C_ * 65];   // pitch 65 to dodge 64-bank conflicts
  int bk = blockIdx.x;
  int k = bk % K_;
  int d = threadIdx.x;
  int kd = k * C_ + d;
  float Ad[NS_];
#pragma unroll
  for (int n = 0; n < NS_; ++n) Ad[n] = -__expf(Alog[(size_t)kd * NS_ + n]);
  float wr[R_];
#pragma unroll
  for (int r = 0; r < R_; ++r) wr[r] = dtw[(size_t)kd * R_ + r];
  float bias = dtb[kd];
  float Dv = Dsp[kd];
  float h[NS_];
#pragma unroll
  for (int n = 0; n < NS_; ++n) h[n] = 0.f;
  const float* xdb = xdbl + (size_t)bk * NR_ * L_;
  const float* xub = xs + (size_t)bk * C_ * L_;
  float* yb = ys + ((size_t)bk * C_ + d) * L_;
  for (int l0 = 0; l0 < L_; l0 += 64) {
    for (int idx = threadIdx.x; idx < NR_ * 64; idx += 96)
      s_xd[idx] = xdb[(size_t)(idx >> 6) * L_ + l0 + (idx & 63)];
    for (int idx = threadIdx.x; idx < C_ * 64; idx += 96)
      s_u[(idx >> 6) * 65 + (idx & 63)] = xub[(size_t)(idx >> 6) * L_ + l0 + (idx & 63)];
    __syncthreads();
    for (int i = 0; i < 64; ++i) {
      float dtraw = bias;
#pragma unroll
      for (int r = 0; r < R_; ++r) dtraw += s_xd[r * 64 + i] * wr[r];
      float dt = softplus_f(dtraw);
      float u = s_u[d * 65 + i];
      float dtu = dt * u;
      float y = 0.f;
#pragma unroll
      for (int n = 0; n < NS_; ++n) {
        float Bv = s_xd[(R_ + n) * 64 + i];
        float Cv = s_xd[(R_ + NS_ + n) * 64 + i];
        h[n] = __expf(dt * Ad[n]) * h[n] + dtu * Bv;
        y += h[n] * Cv;
      }
      yb[l0 + i] = y + u * Dv;
    }
    __syncthreads();
  }
}

// Combine three scan directions, un-permute, transpose to (B,H,W,C)
__global__ void k_combine(const float* __restrict__ ys, const int* __restrict__ inv,
                          float* __restrict__ out) {
  int idx = blockIdx.x * blockDim.x + threadIdx.x;
  if (idx >= B_ * C_ * L_) return;
  int l = idx % L_;
  int bd = idx / L_;
  int d = bd % C_, b = bd / C_;
  size_t base = ((size_t)b * K_) * C_ * L_ + (size_t)d * L_;
  float y0 = ys[base + 0 * (size_t)C_ * L_ + l];
  float y1 = ys[base + 1 * (size_t)C_ * L_ + (L_ - 1 - l)];
  float y2 = ys[base + 2 * (size_t)C_ * L_ + inv[b * L_ + l]];
  out[((size_t)b * L_ + l) * C_ + d] = (y0 + y1 + y2) * (1.f / 3.f);
}

// ---------------- host side ----------------
extern "C" void kernel_launch(void* const* d_in, const int* in_sizes, int n_in,
                              void* d_out, int out_size, void* d_ws, size_t ws_size,
                              hipStream_t stream) {
  const float* x    = (const float*)d_in[0];
  const float* Wp   = (const float*)d_in[1];
  const float* dtw  = (const float*)d_in[2];
  const float* dtb  = (const float*)d_in[3];
  const float* Alog = (const float*)d_in[4];
  const float* Dsp  = (const float*)d_in[5];
  const float* w1   = (const float*)d_in[6];
  const float* b1   = (const float*)d_in[7];
  const float* ca1w = (const float*)d_in[8];
  const float* ca1b = (const float*)d_in[9];
  const float* ca2w = (const float*)d_in[10];
  const float* ca2b = (const float*)d_in[11];
  const float* lng  = (const float*)d_in[12];
  const float* lnb  = (const float*)d_in[13];
  const float* w2   = (const float*)d_in[14];
  const float* rpet = (const float*)d_in[15];
  float* out = (float*)d_out;

  char* p = (char*)d_ws;
  auto carve = [&](size_t bytes) {
    char* r = p;
    p += (bytes + 255) & ~(size_t)255;
    return r;
  };
  float*     xc    = (float*)    carve((size_t)B_ * C_ * 4);
  float*     ca    = (float*)    carve((size_t)B_ * C_ * 4);
  float*     x1    = (float*)    carve((size_t)B_ * C_ * L_ * 4);
  float*     off   = (float*)    carve((size_t)B_ * 3 * L_ * 4);
  float*     rb    = (float*)    carve((size_t)C_ * L_ * 4);
  float*     xd    = (float*)    carve((size_t)B_ * C_ * L_ * 4);
  int*       indc  = (int*)      carve((size_t)B_ * L_ * 4);
  int*       inv   = (int*)      carve((size_t)B_ * L_ * 4);
  float*     xs    = (float*)    carve((size_t)B_ * K_ * C_ * L_ * 4);
  _Float16*  xs_h  = (_Float16*) carve((size_t)B_ * K_ * L_ * C_ * 2);
  _Float16*  Apack = (_Float16*) carve((size_t)K_ * 3 * 3 * 32 * 16 * 2);
  float*     xdbl  = (float*)    carve((size_t)B_ * K_ * NR_ * L_ * 4);
  float*     ys    = (float*)    carve((size_t)B_ * K_ * C_ * L_ * 4);

  k_mean   <<<B_ * C_, 256, 0, stream>>>(x, xc);
  k_ca     <<<(B_ * C_ + 255) / 256, 256, 0, stream>>>(xc, ca1w, ca1b, ca2w, ca2b, ca);
  k_dwconv <<<(B_ * C_ * L_ + 255) / 256, 256, 0, stream>>>(x, w1, b1, ca, x1);
  k_lnoff  <<<(B_ * L_ + 255) / 256, 256, 0, stream>>>(x1, lng, lnb, w2, off);
  k_rpe    <<<(C_ * L_ + 255) / 256, 256, 0, stream>>>(rpet, rb);
  k_sample <<<(B_ * L_ + 255) / 256, 256, 0, stream>>>(x, off, rb, xd);
  k_sort   <<<B_, 1024, 0, stream>>>(off, indc, inv);
  k_buildxs<<<(B_ * C_ * L_ + 255) / 256, 256, 0, stream>>>(x, xd, indc, xs, xs_h);
  k_packA  <<<(K_ * 3 * 3 * 32 + 255) / 256, 256, 0, stream>>>(Wp, Apack);
  k_gemm   <<<B_ * K_ * 3 * (L_ / 64), 32, 0, stream>>>(xs_h, Apack, xdbl);
  k_scan   <<<B_ * K_, 96, 0, stream>>>(xs, xdbl, dtw, dtb, Alog, Dsp, ys);
  k_combine<<<(B_ * C_ * L_ + 255) / 256, 256, 0, stream>>>(ys, inv, out);
}
